// RNNModule_10170482557182
// MI455X (gfx1250) — matrix operand order repaired and checked by reading, not compiled, not run on hardware
//
#include <hip/hip_runtime.h>
#include <hip/hip_bf16.h>
#include <stdint.h>

// MI455X (gfx1250) 2-layer LSTM, latency-bound recurrence.
// Phase A (grid-parallel): pre0[t,b,g] = x_seq[t,b,:]@W_ih0^T + b_ih0 + b_hh0,
//   plus f16 padded weight images for the recurrent matrices in d_ws.
// Phase B (1 persistent workgroup, 14 waves): TDM-preload weights into LDS,
//   pipelined layer0/layer1 recurrences with WMMA f16 GEMMs, + final FC.

#define TT 1024
#define HH 100
#define KP 128   // padded K stride (halfs): no predication anywhere in the hot loop

typedef __attribute__((ext_vector_type(16))) _Float16     v16h;
typedef __attribute__((ext_vector_type(8)))  _Float16     v8h;
typedef __attribute__((ext_vector_type(8)))  float        v8f;
typedef __attribute__((ext_vector_type(4)))  unsigned int v4u;
typedef __attribute__((ext_vector_type(8)))  int          v8i;
typedef __attribute__((ext_vector_type(4)))  int          v4i;

__device__ __forceinline__ v8f wmma_f16(v16h a, v16h b, v8f c) {
    return __builtin_amdgcn_wmma_f32_16x16x32_f16(
        /*neg_a=*/false, a, /*neg_b=*/false, b,
        /*c_mod=*/(short)0, c, /*reuse_a=*/false, /*reuse_b=*/false);
}

// B fragment (32x16, KxN): N = lane&15; lanes 0-15 hold K = kc*32+0..15,
// lanes 16-31 hold K = kc*32+16..31 (16 contiguous halfs per lane).
__device__ __forceinline__ v16h load_bfrag(const _Float16* img, int n0, int kc, int lane) {
    int n = n0 + (lane & 15);
    n = n < 400 ? n : 399;                    // clamp padded gate rows (lanes masked later)
    int kb = (kc << 5) + ((lane & 16) ? 16 : 0);
    return *(const v16h*)(img + n * KP + kb); // 32B aligned, always in range (KP=128)
}

// A fragment (16x32, MxK), contiguous K layout matching B: M = lane&15;
// halfs 0..7  = K kc*32 + (lane<16 ? 0..7  : 16..23)
// halfs 8..15 = K kc*32 + (lane<16 ? 8..15 : 24..31)
__device__ __forceinline__ v16h load_afrag(const _Float16* img, int kc, int lane) {
    int m  = lane & 15;
    int g0 = (kc << 5) + ((lane & 16) ? 16 : 0);
    v8h lo = *(const v8h*)(img + m * KP + g0);        // 16B aligned
    v8h hi = *(const v8h*)(img + m * KP + g0 + 8);
    return __builtin_shufflevector(lo, hi, 0,1,2,3,4,5,6,7,8,9,10,11,12,13,14,15);
}

// ---------------- Phase A1: layer-0 input GEMM (in=3) ----------------
__global__ void lstm_pre0_kernel(const float* __restrict__ x,
                                 const float* __restrict__ Wih0,
                                 const float* __restrict__ bih0,
                                 const float* __restrict__ bhh0,
                                 float* __restrict__ pre0) {
    int idx = blockIdx.x * blockDim.x + threadIdx.x;
    if (idx >= TT * 400) return;
    int g = idx % 400;
    int t = idx / 400;
    float w0 = Wih0[g * 3 + 0], w1 = Wih0[g * 3 + 1], w2 = Wih0[g * 3 + 2];
    float bias = bih0[g] + bhh0[g];
#pragma unroll
    for (int m = 0; m < 3; ++m) {
        // seq[t,b,i] = x[b*3072 + i*1024 + t]
        float x0 = x[m * 3072 + 0    + t];
        float x1 = x[m * 3072 + 1024 + t];
        float x2 = x[m * 3072 + 2048 + t];
        pre0[t * 1200 + m * 400 + g] =
            fmaf(w2, x2, fmaf(w1, x1, fmaf(w0, x0, bias)));
    }
}

// ---------------- Phase A2: padded f16 weight images for TDM ----------------
// wimg[mtx][r][k] (k stride KP): mtx 0=W_hh0, 1=W_ih1, 2=W_hh1; k>=100 zero-padded.
__global__ void lstm_wimg_kernel(const float* __restrict__ Whh0,
                                 const float* __restrict__ Wih1,
                                 const float* __restrict__ Whh1,
                                 _Float16* __restrict__ wimg) {
    int idx = blockIdx.x * blockDim.x + threadIdx.x;   // 3*400*KP elements
    if (idx >= 3 * 400 * KP) return;
    int k   = idx % KP;
    int r   = (idx / KP) % 400;
    int mtx = idx / (400 * KP);
    const float* src = (mtx == 0) ? Whh0 : (mtx == 1) ? Wih1 : Whh1;
    float v = (k < HH) ? src[r * HH + k] : 0.0f;
    wimg[idx] = (_Float16)v;
}

// ---------------- Phase B: persistent pipelined recurrence ----------------
struct alignas(64) Smem {
    _Float16 W[3][400 * KP];     // 307,200 B : filled by TENSOR_LOAD_TO_LDS
    _Float16 H[2][2][16 * KP];   //  16,384 B : [layer][buf] h images (16 x KP)
    float    bias1[400];         //   1,600 B : b_ih1 + b_hh1
    float    hfin[3 * KP];       //   1,536 B : final layer-1 h (f32) for FC
};                               // total 326,720 B <= 320 KB/WGP

__global__ __launch_bounds__(448, 1)
void lstm_recurrent_kernel(const float* __restrict__ pre0,
                           const _Float16* __restrict__ wimg,
                           const float* __restrict__ bih1,
                           const float* __restrict__ bhh1,
                           const float* __restrict__ fcw,
                           const float* __restrict__ fcb,
                           float* __restrict__ out) {
    __shared__ Smem s;
    const int tid   = threadIdx.x;
    const int lane  = tid & 31;
    const int wave  = tid >> 5;
    const int layer = wave / 7;       // waves 0-6: layer0, waves 7-13: layer1
    const int w     = wave % 7;       // hidden slice [16w, 16w+16)

    // ---- TDM: DMA all three f16 weight images global -> LDS (one 2D tile:
    //      256 B rows x 1200 rows, contiguous). Issued by wave 0 only. ----
    if (wave == 0) {
        uint64_t ga      = (uint64_t)(uintptr_t)wimg;
        uint32_t lds_off = (uint32_t)(uintptr_t)(&s.W[0][0]);
        // D# group0: count=1 | lds_addr | global_addr[56:0] | type=2
        v4u g0;
        g0.x = 1u;
        g0.y = lds_off;
        g0.z = (uint32_t)ga;
        g0.w = (uint32_t)((ga >> 32) & 0x1FFFFFFu) | (2u << 30);
        // D# group1: data_size=1B; tensor_dim0=256, tensor_dim1=1200;
        //            tile_dim0=256, tile_dim1=1200; tensor_dim0_stride=256
        v8i g1;
        g1[0] = 0;                 // workgroup_mask=0, data_size=0 (1B), no flags
        g1[1] = (256 << 16);       // abar_addr=0 | tensor_dim0[15:0]
        g1[2] = (1200 << 16);      // tensor_dim0[31:16]=0 | tensor_dim1[15:0]
        g1[3] = (256 << 16);       // tensor_dim1[31:16]=0 | tile_dim0
        g1[4] = 1200;              // tile_dim1 | tile_dim2=0
        g1[5] = 256;               // tensor_dim0_stride[31:0]
        g1[6] = 0;                 // stride0[47:32] | stride1[15:0]
        g1[7] = 0;                 // stride1[47:16]
        v4i gz4 = {0, 0, 0, 0};            // groups 2/3 unused (2-D tile)
        v8i gz8 = {0, 0, 0, 0, 0, 0, 0, 0};
        __builtin_amdgcn_tensor_load_to_lds(g0, g1, gz4, gz4, gz8, 0);
    }

    // ---- init small LDS regions in parallel with the DMA ----
    {
        int* zp = (int*)&s.H[0][0][0];
        for (int i = tid; i < (int)(sizeof(s.H) / 4); i += 448) zp[i] = 0;
        for (int i = tid; i < 400; i += 448) s.bias1[i] = bih1[i] + bhh1[i];
        for (int i = tid; i < 3 * KP; i += 448) s.hfin[i] = 0.0f;
    }
    if (wave == 0) __builtin_amdgcn_s_wait_tensorcnt(0);
    __syncthreads();

    const int  j     = (w << 4) + (lane & 15);               // hidden index this lane owns
    const bool valid = (lane < 16) && (j < HH);
    float c[3]  = {0.f, 0.f, 0.f};
    float hl[3] = {0.f, 0.f, 0.f};
    float b_i = 0.f, b_f = 0.f, b_g = 0.f, b_o = 0.f;
    if (valid) {                                             // layer-1 gate biases
        b_i = s.bias1[j];       b_f = s.bias1[100 + j];
        b_g = s.bias1[200 + j]; b_o = s.bias1[300 + j];
    }
    const _Float16* Wrec = (layer == 0) ? s.W[0] : s.W[2];
    const int n0i = (w << 4), n0f = 100 + (w << 4), n0g = 200 + (w << 4), n0o = 300 + (w << 4);

    for (int it = 0; it <= TT; ++it) {
        const int rb = (it + 1) & 1;                         // read buffer (h[t-1])
        const int wb = it & 1;                               // write buffer (h[t])
        const bool active = (layer == 0) ? (it < TT) : (it >= 1);
        if (active) {
            const int t = (layer == 0) ? it : it - 1;
            v8f a0 = {}, a1 = {}, a2 = {}, a3 = {};
            // recurrent term: h_prev(own layer) @ W_rec^T
            const _Float16* hin = &s.H[layer][rb][0];
#pragma unroll
            for (int kc = 0; kc < 4; ++kc) {
                v16h a = load_afrag(hin, kc, lane);
                a0 = wmma_f16(a, load_bfrag(Wrec, n0i, kc, lane), a0);
                a1 = wmma_f16(a, load_bfrag(Wrec, n0f, kc, lane), a1);
                a2 = wmma_f16(a, load_bfrag(Wrec, n0g, kc, lane), a2);
                a3 = wmma_f16(a, load_bfrag(Wrec, n0o, kc, lane), a3);
            }
            if (layer == 1) {
                // input term: h0[t] @ W_ih1^T  (h0[t] written by layer0 at it-1 -> buf rb)
                const _Float16* h0in = &s.H[0][rb][0];
#pragma unroll
                for (int kc = 0; kc < 4; ++kc) {
                    v16h a = load_afrag(h0in, kc, lane);
                    a0 = wmma_f16(a, load_bfrag(s.W[1], n0i, kc, lane), a0);
                    a1 = wmma_f16(a, load_bfrag(s.W[1], n0f, kc, lane), a1);
                    a2 = wmma_f16(a, load_bfrag(s.W[1], n0g, kc, lane), a2);
                    a3 = wmma_f16(a, load_bfrag(s.W[1], n0o, kc, lane), a3);
                }
            }
            // elementwise LSTM update: lane<16, VGPR m of acc = (batch m, col j)
            if (valid) {
#pragma unroll
                for (int m = 0; m < 3; ++m) {
                    float gi, gf, gg, go;
                    if (layer == 0) {
                        int base = t * 1200 + m * 400 + j;   // pre-activations (L2-resident)
                        gi = a0[m] + pre0[base];
                        gf = a1[m] + pre0[base + 100];
                        gg = a2[m] + pre0[base + 200];
                        go = a3[m] + pre0[base + 300];
                    } else {
                        gi = a0[m] + b_i; gf = a1[m] + b_f;
                        gg = a2[m] + b_g; go = a3[m] + b_o;
                    }
                    float i_ = 1.0f / (1.0f + __expf(-gi));
                    float f_ = 1.0f / (1.0f + __expf(-gf));
                    float g_ = tanhf(gg);
                    float o_ = 1.0f / (1.0f + __expf(-go));
                    float cn = f_ * c[m] + i_ * g_;
                    c[m] = cn;
                    float hn = o_ * tanhf(cn);
                    hl[m] = hn;
                    s.H[layer][wb][m * KP + j] = (_Float16)hn;
                }
            }
        }
        __syncthreads();   // one barrier per pipeline step
    }

    // ---- epilogue: FC on final layer-1 hidden state ----
    if (layer == 1 && valid) {
#pragma unroll
        for (int m = 0; m < 3; ++m) s.hfin[m * KP + j] = hl[m];
    }
    __syncthreads();
    if (tid < 30) {
        int k = tid % 10, m = tid / 10;
        float sum = fcb[k];
        for (int jj = 0; jj < HH; ++jj)
            sum = fmaf(s.hfin[m * KP + jj], fcw[k * HH + jj], sum);
        out[m * 10 + k] = sum;
    }
}

extern "C" void kernel_launch(void* const* d_in, const int* in_sizes, int n_in,
                              void* d_out, int out_size, void* d_ws, size_t ws_size,
                              hipStream_t stream) {
    const float* x     = (const float*)d_in[0];
    const float* W_ih0 = (const float*)d_in[1];
    const float* W_hh0 = (const float*)d_in[2];
    const float* b_ih0 = (const float*)d_in[3];
    const float* b_hh0 = (const float*)d_in[4];
    const float* W_ih1 = (const float*)d_in[5];
    const float* W_hh1 = (const float*)d_in[6];
    const float* b_ih1 = (const float*)d_in[7];
    const float* b_hh1 = (const float*)d_in[8];
    const float* fc_w  = (const float*)d_in[9];
    const float* fc_b  = (const float*)d_in[10];

    float*    pre0 = (float*)d_ws;                           // 1024*1200 f32 = 4,915,200 B
    _Float16* wimg = (_Float16*)((char*)d_ws + (size_t)TT * 1200 * 4); // 307,200 B

    lstm_pre0_kernel<<<(TT * 400 + 255) / 256, 256, 0, stream>>>(
        x, W_ih0, b_ih0, b_hh0, pre0);
    lstm_wimg_kernel<<<(3 * 400 * KP + 255) / 256, 256, 0, stream>>>(
        W_hh0, W_ih1, W_hh1, wimg);
    lstm_recurrent_kernel<<<1, 448, 0, stream>>>(
        pre0, wimg, b_ih1, b_hh1, fc_w, fc_b, (float*)d_out);
}
